// TransformerBlock_1872605741671
// MI455X (gfx1250) — matrix operand order repaired and checked by reading, not compiled, not run on hardware
//
#include <hip/hip_runtime.h>
#include <cstdint>
#include <cstddef>

// ---------------------------------------------------------------------------
// Types
// ---------------------------------------------------------------------------
typedef _Float16 f16;
typedef __attribute__((ext_vector_type(16))) _Float16 v16h;
typedef __attribute__((ext_vector_type(4)))  _Float16 v4h;
typedef __attribute__((ext_vector_type(8)))  float    v8f;
typedef __attribute__((ext_vector_type(4)))  int      v4i;

#define E_DIM   1024
#define H_NUM   16
#define D_HEAD  64
#define FF_DIM  4096
#define B_SZ    2
#define L_SEQ   2048
#define M_ROWS  (B_SZ * L_SEQ)   // 4096 tokens

// gfx1250 async global->LDS path (ASYNCcnt), if this toolchain exposes it
#if defined(__has_builtin)
#if __has_builtin(__builtin_amdgcn_global_load_async_to_lds_b128)
#define HAS_ASYNC_LDS 1
#endif
#endif
#ifndef HAS_ASYNC_LDS
#define HAS_ASYNC_LDS 0
#endif

// builtin signature: (int4 AS(1)*, int4 AS(3)*, imm offset, imm cpol)
#define TO_GLOBAL(p) ((__attribute__((address_space(1))) v4i*)(p))
#define TO_LDS(p)    ((__attribute__((address_space(3))) v4i*)(p))

static __device__ __forceinline__ void wait_async0() {
#if defined(__has_builtin)
#if __has_builtin(__builtin_amdgcn_s_wait_asynccnt)
  __builtin_amdgcn_s_wait_asynccnt(0);
#else
  asm volatile("s_wait_asynccnt 0x0" ::: "memory");
#endif
#else
  asm volatile("s_wait_asynccnt 0x0" ::: "memory");
#endif
}

static __device__ __forceinline__ v8f wmma16(v16h a, v16h b, v8f c) {
  // D(f32 16x16) = A(f16 16x32) * B(f16 32x16) + C
  return __builtin_amdgcn_wmma_f32_16x16x32_f16(false, a, false, b, (short)0, c,
                                                false, false);
}

// Gather a 16x32 f16 A-fragment (row-major source, given row stride in f16).
// Layout: lane half hl, element i holds K = 16*hl + i (contiguous, matching B).
// Loaded as 4x 8-byte chunks.
static __device__ __forceinline__ v16h load_afrag(const f16* __restrict__ base,
                                                  int stride, int m, int hl) {
  union { v16h v; uint2 q[4]; } t;
  const f16* row = base + (size_t)m * stride;
#pragma unroll
  for (int jj = 0; jj < 4; ++jj) {
    int kb = 16 * hl + 4 * jj;
    t.q[jj] = *(const uint2*)(row + kb);
  }
  return t.v;
}

// ---------------------------------------------------------------------------
// fp32 -> f16 conversion
// ---------------------------------------------------------------------------
__global__ __launch_bounds__(256) void cvt_f32_f16(const float* __restrict__ src,
                                                   f16* __restrict__ dst, int n4) {
  int i = blockIdx.x * 256 + threadIdx.x;
  if (i < n4) {
    float4 v = ((const float4*)src)[i];
    v4h o = { (f16)v.x, (f16)v.y, (f16)v.z, (f16)v.w };
    ((v4h*)dst)[i] = o;
  }
}

// ---------------------------------------------------------------------------
// WMMA GEMM: C[M,N] = A[M,K] * W[N,K]^T  (+bias, +gelu, +residual) per flags
// Block tile 128x128, 8 waves, wave tile 32x64, K-step 32.
// Async path: double-buffered LDS, GLOBAL_LOAD_ASYNC_TO_LDS_B128 fills buffer
// k+1 while WMMAs consume buffer k. Fallback: register-prefetch staging.
// ---------------------------------------------------------------------------
template <bool OUT16, bool BIASF, bool GELUF, bool RESF>
__global__ __launch_bounds__(256) void gemm_wmma(
    const f16* __restrict__ A, const f16* __restrict__ W,
    const float* __restrict__ bias, const float* __restrict__ res,
    f16* __restrict__ out16, float* __restrict__ out32,
    int M, int N, int K) {
  const int tid  = threadIdx.x;
  const int wave = tid >> 5;
  const int lane = tid & 31;
  const int hl   = lane >> 4;
  const int ln   = lane & 15;
  const int m0 = blockIdx.y * 128;
  const int n0 = blockIdx.x * 128;
  const int wm = wave & 3;   // 4 waves along M (32 rows each)
  const int wn = wave >> 2;  // 2 waves along N (64 cols each)

  v8f acc[2][4];
#pragma unroll
  for (int mt = 0; mt < 2; ++mt)
#pragma unroll
    for (int nt = 0; nt < 4; ++nt) acc[mt][nt] = (v8f){0.f,0.f,0.f,0.f,0.f,0.f,0.f,0.f};

  const int KS = K >> 5;
  const f16* Acur;
  const f16* Bcur;

#if HAS_ASYNC_LDS
  __shared__ __attribute__((aligned(32))) f16 As[2][128 * 32];
  __shared__ __attribute__((aligned(32))) f16 Bs[2][128 * 32];

  // each thread owns 2 x 16B chunks of A and of B per tile
  auto issue = [&](int ks, int buf) {
    int kk = ks * 32;
#pragma unroll
    for (int i = 0; i < 2; ++i) {
      int c = tid + i * 256;
      int row = c >> 2, col = (c & 3) * 8;
      __builtin_amdgcn_global_load_async_to_lds_b128(
          TO_GLOBAL(A + (size_t)(m0 + row) * K + kk + col),
          TO_LDS(&As[buf][row * 32 + col]), 0, 0);
      __builtin_amdgcn_global_load_async_to_lds_b128(
          TO_GLOBAL(W + (size_t)(n0 + row) * K + kk + col),
          TO_LDS(&Bs[buf][row * 32 + col]), 0, 0);
    }
  };

  issue(0, 0);
  wait_async0();
  __syncthreads();
#else
  __shared__ __attribute__((aligned(32))) f16 As[1][128 * 32];
  __shared__ __attribute__((aligned(32))) f16 Bs[1][128 * 32];

  auto g_tile = [&](int ks, uint4* pa, uint4* pb) {
    int kk = ks * 32;
#pragma unroll
    for (int i = 0; i < 2; ++i) {
      int c = tid + i * 256;
      int row = c >> 2, col = (c & 3) * 8;
      pa[i] = *(const uint4*)(A + (size_t)(m0 + row) * K + kk + col);
      pb[i] = *(const uint4*)(W + (size_t)(n0 + row) * K + kk + col);
    }
  };
  auto s_tile = [&](const uint4* pa, const uint4* pb) {
#pragma unroll
    for (int i = 0; i < 2; ++i) {
      int c = tid + i * 256;
      int row = c >> 2, col = (c & 3) * 8;
      *(uint4*)(&As[0][row * 32 + col]) = pa[i];
      *(uint4*)(&Bs[0][row * 32 + col]) = pb[i];
    }
  };

  uint4 pa[2], pb[2];
  g_tile(0, pa, pb);
  s_tile(pa, pb);
  __syncthreads();
#endif

  for (int ks = 0; ks < KS; ++ks) {
    const bool more = (ks + 1) < KS;
#if HAS_ASYNC_LDS
    const int cur = ks & 1;
    if (more) issue(ks + 1, cur ^ 1);  // async fill of next buffer during WMMAs
    Acur = &As[cur][0];
    Bcur = &Bs[cur][0];
#else
    uint4 na[2], nb[2];
    if (more) g_tile(ks + 1, na, nb);  // overlap global loads with WMMAs
    Acur = &As[0][0];
    Bcur = &Bs[0][0];
#endif

    v16h af[2], bf[4];
#pragma unroll
    for (int mt = 0; mt < 2; ++mt)
      af[mt] = load_afrag(Acur, 32, wm * 32 + mt * 16 + ln, hl);
#pragma unroll
    for (int nt = 0; nt < 4; ++nt) {
      // B 32x16 fragment: lane column n, 16 contiguous K per lane-half
      const f16* p = Bcur + (size_t)(wn * 64 + nt * 16 + ln) * 32 + 16 * hl;
      bf[nt] = *(const v16h*)p;
    }
#pragma unroll
    for (int mt = 0; mt < 2; ++mt)
#pragma unroll
      for (int nt = 0; nt < 4; ++nt)
        acc[mt][nt] = wmma16(af[mt], bf[nt], acc[mt][nt]);

#if HAS_ASYNC_LDS
    if (more) wait_async0();
    __syncthreads();
#else
    __syncthreads();
    if (more) s_tile(na, nb);
    __syncthreads();
#endif
  }

  // Epilogue: C-layout rows M = hl*8 + r, cols N = ln
#pragma unroll
  for (int mt = 0; mt < 2; ++mt) {
#pragma unroll
    for (int nt = 0; nt < 4; ++nt) {
#pragma unroll
      for (int r = 0; r < 8; ++r) {
        int row = m0 + wm * 32 + mt * 16 + hl * 8 + r;
        int col = n0 + wn * 64 + nt * 16 + ln;
        float v = acc[mt][nt][r];
        if (BIASF) v += bias[col];
        if (GELUF) v = 0.5f * v * (1.0f + erff(v * 0.70710678118654752f));
        size_t idx = (size_t)row * N + col;
        if (RESF) v += res[idx];
        if (OUT16) out16[idx] = (f16)v;
        else       out32[idx] = v;
      }
    }
  }
}

// ---------------------------------------------------------------------------
// Flash attention: one block per (b, h, 128 query rows); wave = 16 queries.
// Key tiles of 32, online softmax, all matmuls via WMMA. K tile goes through
// the async global->LDS path when available; V needs a transpose so it stays
// on the register path.
// ---------------------------------------------------------------------------
__global__ __launch_bounds__(256) void attn_flash(const f16* __restrict__ Q,
                                                  const f16* __restrict__ Km,
                                                  const f16* __restrict__ Vm,
                                                  f16* __restrict__ Ctx) {
  __shared__ __attribute__((aligned(32))) f16 Ks[32 * 64];      // [key, d]
  __shared__ __attribute__((aligned(32))) f16 Vt[64 * 32];      // [d, key]
  __shared__ __attribute__((aligned(32))) f16 Pb[8][16 * 32];   // per-wave P

  const int tid  = threadIdx.x;
  const int wave = tid >> 5;
  const int lane = tid & 31;
  const int hl   = lane >> 4;
  const int ln   = lane & 15;
  const int b = blockIdx.y / H_NUM;
  const int h = blockIdx.y % H_NUM;
  const int q0 = blockIdx.x * 128;
  const size_t headoff = (size_t)b * L_SEQ * E_DIM + (size_t)h * D_HEAD;

  // Q fragments (16 rows x 64 d = two K-steps), resident for whole loop
  v16h qf[2];
  {
    const f16* qrow = Q + headoff + (size_t)(q0 + wave * 16 + ln) * E_DIM;
#pragma unroll
    for (int ds = 0; ds < 2; ++ds) {
      union { v16h v; uint2 u[4]; } t;
#pragma unroll
      for (int jj = 0; jj < 4; ++jj) {
        int kb = ds * 32 + 16 * hl + 4 * jj;
        t.u[jj] = *(const uint2*)(qrow + kb);
      }
      qf[ds] = t.v;
    }
  }

  float mrow[8], lrow[8];
  v8f O[4];
#pragma unroll
  for (int r = 0; r < 8; ++r) { mrow[r] = -1e30f; lrow[r] = 0.f; }
#pragma unroll
  for (int dt = 0; dt < 4; ++dt) O[dt] = (v8f){0.f,0.f,0.f,0.f,0.f,0.f,0.f,0.f};

  const int row_c = tid >> 3;        // 0..31  (key row)
  const int col_c = (tid & 7) * 8;   // 0..56  (d offset)

  for (int kt = 0; kt < L_SEQ / 32; ++kt) {
    const int kbase = kt * 32;
    {  // cooperative K/V tile load; V stored transposed for the P*V B-frags
      const f16* ksrc = Km + headoff + (size_t)(kbase + row_c) * E_DIM + col_c;
#if HAS_ASYNC_LDS
      __builtin_amdgcn_global_load_async_to_lds_b128(
          TO_GLOBAL(ksrc), TO_LDS(Ks + row_c * 64 + col_c), 0, 0);
#else
      *(uint4*)(Ks + row_c * 64 + col_c) = *(const uint4*)ksrc;
#endif
      union { uint4 u; f16 hh[8]; } vv;
      vv.u = *(const uint4*)(Vm + headoff + (size_t)(kbase + row_c) * E_DIM + col_c);
#pragma unroll
      for (int j = 0; j < 8; ++j) Vt[(col_c + j) * 32 + row_c] = vv.hh[j];
#if HAS_ASYNC_LDS
      wait_async0();
#endif
    }
    __syncthreads();

    // scores S(16x32) = Q(16x64) * K^T(64x32), 4 WMMAs
    v8f S[2];
    S[0] = (v8f){0.f,0.f,0.f,0.f,0.f,0.f,0.f,0.f};
    S[1] = (v8f){0.f,0.f,0.f,0.f,0.f,0.f,0.f,0.f};
#pragma unroll
    for (int ds = 0; ds < 2; ++ds) {
#pragma unroll
      for (int nt = 0; nt < 2; ++nt) {
        const f16* p = Ks + (size_t)(nt * 16 + ln) * 64 + ds * 32 + 16 * hl;
        v16h kf = *(const v16h*)p;
        S[nt] = wmma16(qf[ds], kf, S[nt]);
      }
    }

    // online softmax; row M = hl*8 + r lives in VGPR r across a 16-lane half
    float alpha[8];
#pragma unroll
    for (int r = 0; r < 8; ++r) {
      float s0 = S[0][r] * 0.125f;  // 1/sqrt(64)
      float s1 = S[1][r] * 0.125f;
      float mx = fmaxf(s0, s1);
#pragma unroll
      for (int off = 8; off >= 1; off >>= 1) mx = fmaxf(mx, __shfl_xor(mx, off, 32));
      float mn = fmaxf(mrow[r], mx);
      float al = __expf(mrow[r] - mn);
      float p0 = __expf(s0 - mn);
      float p1 = __expf(s1 - mn);
      float ps = p0 + p1;
#pragma unroll
      for (int off = 8; off >= 1; off >>= 1) ps += __shfl_xor(ps, off, 32);
      lrow[r] = lrow[r] * al + ps;
      mrow[r] = mn;
      alpha[r] = al;
      S[0][r] = p0;
      S[1][r] = p1;
    }
#pragma unroll
    for (int dt = 0; dt < 4; ++dt)
#pragma unroll
      for (int r = 0; r < 8; ++r) O[dt][r] *= alpha[r];

    // P: C-layout -> LDS -> A-fragment (per-wave region, DS ops are in-order)
    f16* Pw = Pb[wave];
#pragma unroll
    for (int nt = 0; nt < 2; ++nt)
#pragma unroll
      for (int r = 0; r < 8; ++r)
        Pw[(hl * 8 + r) * 32 + nt * 16 + ln] = (f16)S[nt][r];

    v16h pf = load_afrag(Pw, 32, ln, hl);

    // O(16x64) += P(16x32) * V(32x64), 4 WMMAs
#pragma unroll
    for (int dt = 0; dt < 4; ++dt) {
      v16h vf = *(const v16h*)(Vt + (size_t)(dt * 16 + ln) * 32 + 16 * hl);
      O[dt] = wmma16(pf, vf, O[dt]);
    }
    __syncthreads();
  }

  // normalize and write context (f16, (B,L,E) layout)
#pragma unroll
  for (int dt = 0; dt < 4; ++dt) {
#pragma unroll
    for (int r = 0; r < 8; ++r) {
      int m = q0 + wave * 16 + hl * 8 + r;
      float v = O[dt][r] / lrow[r];
      Ctx[headoff + (size_t)m * E_DIM + dt * 16 + ln] = (f16)v;
    }
  }
}

// ---------------------------------------------------------------------------
// LayerNorm: one block per row of 1024; optional f16 copy of the output
// ---------------------------------------------------------------------------
template <bool W16>
__global__ __launch_bounds__(256) void layernorm_k(const float* __restrict__ in,
                                                   const float* __restrict__ g,
                                                   const float* __restrict__ bt,
                                                   float* __restrict__ out32,
                                                   f16* __restrict__ out16) {
  __shared__ float red[16];
  const int tid = threadIdx.x;
  const size_t base = (size_t)blockIdx.x * E_DIM;
  float4 x = ((const float4*)(in + base))[tid];
  float s  = x.x + x.y + x.z + x.w;
  float s2 = x.x * x.x + x.y * x.y + x.z * x.z + x.w * x.w;
#pragma unroll
  for (int off = 16; off >= 1; off >>= 1) {
    s  += __shfl_xor(s, off, 32);
    s2 += __shfl_xor(s2, off, 32);
  }
  if ((tid & 31) == 0) { red[tid >> 5] = s; red[8 + (tid >> 5)] = s2; }
  __syncthreads();
  float ts = 0.f, ts2 = 0.f;
#pragma unroll
  for (int i = 0; i < 8; ++i) { ts += red[i]; ts2 += red[8 + i]; }
  float mean = ts * (1.0f / E_DIM);
  float var  = ts2 * (1.0f / E_DIM) - mean * mean;
  float rs   = rsqrtf(var + 1e-5f);
  float4 gg = ((const float4*)g)[tid];
  float4 bb = ((const float4*)bt)[tid];
  float4 y;
  y.x = (x.x - mean) * rs * gg.x + bb.x;
  y.y = (x.y - mean) * rs * gg.y + bb.y;
  y.z = (x.z - mean) * rs * gg.z + bb.z;
  y.w = (x.w - mean) * rs * gg.w + bb.w;
  ((float4*)(out32 + base))[tid] = y;
  if (W16) {
    v4h o = { (f16)y.x, (f16)y.y, (f16)y.z, (f16)y.w };
    ((v4h*)(out16 + base))[tid] = o;
  }
}

// ---------------------------------------------------------------------------
// Launch
// ---------------------------------------------------------------------------
extern "C" void kernel_launch(void* const* d_in, const int* in_sizes, int n_in,
                              void* d_out, int out_size, void* d_ws, size_t ws_size,
                              hipStream_t stream) {
  (void)in_sizes; (void)n_in; (void)out_size; (void)ws_size;
  const float* x   = (const float*)d_in[0];
  const float* wq  = (const float*)d_in[1];
  const float* wk  = (const float*)d_in[2];
  const float* wv  = (const float*)d_in[3];
  const float* wo  = (const float*)d_in[4];
  const float* bo  = (const float*)d_in[5];
  const float* g1  = (const float*)d_in[6];
  const float* b1  = (const float*)d_in[7];
  const float* w1  = (const float*)d_in[8];
  const float* bf1 = (const float*)d_in[9];
  const float* w2  = (const float*)d_in[10];
  const float* bf2 = (const float*)d_in[11];
  const float* g2  = (const float*)d_in[12];
  const float* b2  = (const float*)d_in[13];

  char* ws = (char*)d_ws;
  const size_t MB = 1024u * 1024u;
  f16*   x16   = (f16*)(ws + 0 * MB);    // 8MB, reused as h16 later
  f16*   wq16  = (f16*)(ws + 8 * MB);    // 2MB
  f16*   wk16  = (f16*)(ws + 10 * MB);   // 2MB
  f16*   wv16  = (f16*)(ws + 12 * MB);   // 2MB
  f16*   wo16  = (f16*)(ws + 14 * MB);   // 2MB
  f16*   w116  = (f16*)(ws + 16 * MB);   // 8MB
  f16*   w216  = (f16*)(ws + 24 * MB);   // 8MB
  f16*   q16   = (f16*)(ws + 32 * MB);   // 8MB  (region 32..64MB reused as ff1)
  f16*   k16   = (f16*)(ws + 40 * MB);   // 8MB
  f16*   v16   = (f16*)(ws + 48 * MB);   // 8MB
  f16*   ctx16 = (f16*)(ws + 56 * MB);   // 8MB
  float* pre32 = (float*)(ws + 64 * MB); // 16MB (pre-LN buffer, reused)
  float* h32   = (float*)(ws + 80 * MB); // 16MB
  f16*   h16   = x16;                    // reuse (x16 dead after QKV GEMMs)
  f16*   ff116 = q16;                    // reuse (q/k/v/ctx dead after O-proj)

  // fp32 -> f16 conversions
  cvt_f32_f16<<<(M_ROWS * E_DIM / 4) / 256, 256, 0, stream>>>(x,  x16,  M_ROWS * E_DIM / 4);
  cvt_f32_f16<<<(E_DIM * E_DIM / 4) / 256, 256, 0, stream>>>(wq, wq16, E_DIM * E_DIM / 4);
  cvt_f32_f16<<<(E_DIM * E_DIM / 4) / 256, 256, 0, stream>>>(wk, wk16, E_DIM * E_DIM / 4);
  cvt_f32_f16<<<(E_DIM * E_DIM / 4) / 256, 256, 0, stream>>>(wv, wv16, E_DIM * E_DIM / 4);
  cvt_f32_f16<<<(E_DIM * E_DIM / 4) / 256, 256, 0, stream>>>(wo, wo16, E_DIM * E_DIM / 4);
  cvt_f32_f16<<<(FF_DIM * E_DIM / 4) / 256, 256, 0, stream>>>(w1, w116, FF_DIM * E_DIM / 4);
  cvt_f32_f16<<<(E_DIM * FF_DIM / 4) / 256, 256, 0, stream>>>(w2, w216, E_DIM * FF_DIM / 4);

  dim3 gE(E_DIM / 128, M_ROWS / 128);   // (8, 32)
  dim3 gF(FF_DIM / 128, M_ROWS / 128);  // (32, 32)

  // Q,K,V projections (f16 out, no bias)
  gemm_wmma<true, false, false, false><<<gE, 256, 0, stream>>>(
      x16, wq16, nullptr, nullptr, q16, nullptr, M_ROWS, E_DIM, E_DIM);
  gemm_wmma<true, false, false, false><<<gE, 256, 0, stream>>>(
      x16, wk16, nullptr, nullptr, k16, nullptr, M_ROWS, E_DIM, E_DIM);
  gemm_wmma<true, false, false, false><<<gE, 256, 0, stream>>>(
      x16, wv16, nullptr, nullptr, v16, nullptr, M_ROWS, E_DIM, E_DIM);

  // attention
  attn_flash<<<dim3(L_SEQ / 128, B_SZ * H_NUM), 256, 0, stream>>>(q16, k16, v16, ctx16);

  // output projection + bias + residual(x) -> pre-LN1 (fp32)
  gemm_wmma<false, true, false, true><<<gE, 256, 0, stream>>>(
      ctx16, wo16, bo, x, nullptr, pre32, M_ROWS, E_DIM, E_DIM);

  // LN1 -> h (fp32 + f16)
  layernorm_k<true><<<M_ROWS, 256, 0, stream>>>(pre32, g1, b1, h32, h16);

  // FFN1: gelu(h @ w1^T + bf1) -> f16
  gemm_wmma<true, true, true, false><<<gF, 256, 0, stream>>>(
      h16, w116, bf1, nullptr, ff116, nullptr, M_ROWS, FF_DIM, E_DIM);

  // FFN2: ff @ w2^T + bf2 + residual(h) -> pre-LN2 (fp32)
  gemm_wmma<false, true, false, true><<<gE, 256, 0, stream>>>(
      ff116, w216, bf2, h32, nullptr, pre32, M_ROWS, E_DIM, FF_DIM);

  // LN2 -> output (fp32)
  layernorm_k<false><<<M_ROWS, 256, 0, stream>>>(pre32, g2, b2, (float*)d_out, nullptr);
}
